// EdgeAwareGCNVEncoder_48507360641532
// MI455X (gfx1250) — matrix operand order repaired and checked by reading, not compiled, not run on hardware
//
#include <hip/hip_runtime.h>

// ---------------------------------------------------------------------------
// EdgeAwareGCNVEncoder for MI455X (gfx1250, wave32, WMMA)
//   dense GEMMs  -> v_wmma_f32_16x16x32_bf16, fragments read from LDS as
//                   contiguous 32B (2x ds_read_b128 per fragment) via
//                   permuted-K A staging + pre-transposed bf16 weights.
//                   All fragments of a K-substep are loaded into distinct
//                   registers before the 4 WMMAs so the scheduler can issue
//                   progressive s_wait_dscnt and overlap LDS with the MFU.
//   edge passes  -> one wave32 per edge, inline 32->HW edge embedding,
//                   segment-softmax via order-preserving uint atomicMax
// ---------------------------------------------------------------------------

typedef __attribute__((ext_vector_type(16))) __bf16 v16bf;
typedef __attribute__((ext_vector_type(8)))  float  v8f;

union FragBF { v16bf v; uint4 q[2]; unsigned short u[16]; };

__device__ __forceinline__ unsigned short bf16_of(float f) {
  unsigned u = __float_as_uint(f);
  unsigned lsb = (u >> 16) & 1u;
  u += 0x7FFFu + lsb;                 // round-to-nearest-even
  return (unsigned short)(u >> 16);
}

// order-preserving fp32 <-> uint encoding (for atomicMax-based segment max)
__device__ __forceinline__ unsigned enc_f32(float f) {
  unsigned u = __float_as_uint(f);
  return (u & 0x80000000u) ? ~u : (u | 0x80000000u);
}
__device__ __forceinline__ float dec_f32(unsigned u) {
  return (u & 0x80000000u) ? __uint_as_float(u & 0x7FFFFFFFu)
                           : __uint_as_float(~u);
}
#define ENC_NEG_INF 0x007FFFFFu  // enc(-inf)

// ---------------------------------------------------------------------------
// elementwise helpers
// ---------------------------------------------------------------------------
__global__ void k_f32_to_bf16(const float* __restrict__ in,
                              unsigned short* __restrict__ out, long n) {
  long i = (long)blockIdx.x * blockDim.x + threadIdx.x;
  long stride = (long)gridDim.x * blockDim.x;
  for (; i < n; i += stride) out[i] = bf16_of(in[i]);
}

// W[k*N + n] (f32) -> Wt[n*K + k] (bf16): weights stored column-major for GEMM
__global__ void k_transpose_bf16(const float* __restrict__ W,
                                 unsigned short* __restrict__ Wt,
                                 int K, int N) {
  long i = (long)blockIdx.x * blockDim.x + threadIdx.x;
  long n = (long)K * N;
  long stride = (long)gridDim.x * blockDim.x;
  for (; i < n; i += stride) {
    int k = (int)(i / N);
    int c = (int)(i - (long)k * N);
    Wt[(size_t)c * K + k] = bf16_of(W[i]);
  }
}

__global__ void k_fill_u32(unsigned* __restrict__ p, unsigned val, long n) {
  long i = (long)blockIdx.x * blockDim.x + threadIdx.x;
  long stride = (long)gridDim.x * blockDim.x;
  for (; i < n; i += stride) p[i] = val;
}

// ---------------------------------------------------------------------------
// bf16 WMMA GEMM: C[M,N] = A[M,K] @ B + bias, optional relu.
//   A  row-major bf16 [M,K]
//   Bt transposed bf16 [N,K]   (weights pre-transposed on device)
// N%64==0, K%64==0, M arbitrary.
// block = 128 threads (4 waves); tile 64(M) x 64(N); BK=64.
// A tile staged with the CDNA5 A-fragment K-permutation so every fragment is
// a contiguous 32B LDS read (2x ds_read_b128).  LDS rows padded to 72 u16
// (144B) -> keeps 16B alignment, staggers banks across the 16 row readers.
// ---------------------------------------------------------------------------
__global__ void k_gemm_bf16(const unsigned short* __restrict__ A,
                            const unsigned short* __restrict__ Bt,
                            const float* __restrict__ bias,
                            float* __restrict__ C,
                            unsigned short* __restrict__ Cbf,
                            int M, int N, int K, int relu) {
  __shared__ __align__(16) unsigned short As[64][72];  // [row][2grp x 32 perm-K]
  __shared__ __align__(16) unsigned short Bs[64][72];  // [col n][2grp x 32 perm-K]

  const int tid  = threadIdx.x;
  const int lane = tid & 31;
  const int w    = tid >> 5;            // wave id 0..3 -> M strip
  const int g    = lane >> 4;           // half-wave 0/1
  const int hl   = lane & 15;

  const int m0 = blockIdx.y * 64;
  const int n0 = blockIdx.x * 64;

  v8f acc[4] = {{}, {}, {}, {}};

  const int srow = tid >> 1;            // staging row 0..63
  const int sgrp = tid & 1;             // which 32-K group

  for (int kk = 0; kk < K; kk += 64) {
    // ---- stage A tile (64 rows x 64 K), permuted within each 32-K group ----
    {
      int gr = m0 + srow;
      uint4 c0 = make_uint4(0, 0, 0, 0), c1 = c0, c2 = c0, c3 = c0;
      if (gr < M) {
        const uint4* gp = (const uint4*)(A + (size_t)gr * K + kk + sgrp * 32);
        c0 = gp[0]; c1 = gp[1]; c2 = gp[2]; c3 = gp[3];
      }
      unsigned short* base = &As[srow][sgrp * 32];
      *(uint4*)(base + 0)  = c0;   // K 0..7   -> frag pos 0..7   (g=0 lo)
      *(uint4*)(base + 16) = c1;   // K 8..15  -> frag pos 16..23 (g=1 lo)
      *(uint4*)(base + 8)  = c2;   // K 16..23 -> frag pos 8..15  (g=0 hi)
      *(uint4*)(base + 24) = c3;   // K 24..31 -> frag pos 24..31 (g=1 hi)
    }
    // ---- stage B tile (64 cols x 64 K), same per-32-K permutation as A --
    {
      const uint4* gp =
          (const uint4*)(Bt + (size_t)(n0 + srow) * K + kk + sgrp * 32);
      unsigned short* base = &Bs[srow][sgrp * 32];
      *(uint4*)(base + 0)  = gp[0];
      *(uint4*)(base + 16) = gp[1];
      *(uint4*)(base + 8)  = gp[2];
      *(uint4*)(base + 24) = gp[3];
    }
    __syncthreads();

#pragma unroll
    for (int ks = 0; ks < 2; ++ks) {
      // issue ALL fragment loads for this K-substep into distinct registers
      // first, then the 4 WMMAs: DS returns are in-order, so the scheduler
      // can use progressive s_wait_dscnt instead of waiting to 0 per WMMA.
      FragBF fa;
      FragBF fb[4];
      {
        const uint4* ap = (const uint4*)&As[w * 16 + hl][ks * 32 + g * 16];
        fa.q[0] = ap[0]; fa.q[1] = ap[1];
      }
#pragma unroll
      for (int sub = 0; sub < 4; ++sub) {
        const uint4* bp = (const uint4*)&Bs[sub * 16 + hl][ks * 32 + g * 16];
        fb[sub].q[0] = bp[0]; fb[sub].q[1] = bp[1];
      }
#pragma unroll
      for (int sub = 0; sub < 4; ++sub) {
        acc[sub] = __builtin_amdgcn_wmma_f32_16x16x32_bf16(
            false, fa.v, false, fb[sub].v, (short)0, acc[sub], false, false);
      }
    }
    __syncthreads();
  }

  // ---- epilogue: bias (+relu) -> f32 (+ optional bf16 mirror) ----
#pragma unroll
  for (int sub = 0; sub < 4; ++sub) {
    int gcol = n0 + sub * 16 + hl;
    float bb = bias[gcol];
#pragma unroll
    for (int j = 0; j < 8; ++j) {
      int grow = m0 + w * 16 + g * 8 + j;
      if (grow < M) {
        float val = acc[sub][j] + bb;
        if (relu) val = fmaxf(val, 0.0f);
        C[(size_t)grow * N + gcol] = val;
        if (Cbf) Cbf[(size_t)grow * N + gcol] = bf16_of(val);
      }
    }
  }
}

// ---------------------------------------------------------------------------
// edge pass 1: raw attention logits + segment max (one wave per edge)
//   alpha[e,h] = dot(q[dst], k[src] + eattr[e]@We + be) / sqrt(C)
// ---------------------------------------------------------------------------
template <int HW, int HEADS>
__global__ void k_edge_alpha(const int* __restrict__ ei,
                             const float* __restrict__ eattr,
                             const float* __restrict__ We,
                             const float* __restrict__ be,
                             const float* __restrict__ q,
                             const float* __restrict__ k,
                             float* __restrict__ alpha,
                             unsigned* __restrict__ amax,
                             int E, float scale) {
  const int CH    = HW / 32;
  const int GROUP = 32 / HEADS;      // lanes per head
  int wave = blockIdx.x * (blockDim.x >> 5) + (threadIdx.x >> 5);
  int lane = threadIdx.x & 31;
  if (wave >= E) return;

  int src = ei[wave];
  int dst = ei[E + wave];
  int c0  = lane * CH;

  float ea = eattr[(size_t)wave * 32 + lane];   // ED == 32
  float emb[CH];
#pragma unroll
  for (int i = 0; i < CH; ++i) emb[i] = be[c0 + i];
#pragma unroll
  for (int j = 0; j < 32; ++j) {
    float aj = __shfl(ea, j, 32);
#pragma unroll
    for (int i = 0; i < CH; ++i) emb[i] += aj * We[j * HW + c0 + i];
  }

  float part = 0.0f;
#pragma unroll
  for (int i = 0; i < CH; ++i) {
    float qd = q[(size_t)dst * HW + c0 + i];
    float ks = k[(size_t)src * HW + c0 + i];
    part += qd * (ks + emb[i]);
  }
#pragma unroll
  for (int off = GROUP / 2; off > 0; off >>= 1)
    part += __shfl_xor(part, off, 32);

  if ((lane & (GROUP - 1)) == 0) {
    int h = lane / GROUP;
    float a = part * scale;
    alpha[(size_t)wave * HEADS + h] = a;
    atomicMax(&amax[(size_t)dst * HEADS + h], enc_f32(a));
  }
}

// ---------------------------------------------------------------------------
// edge pass 2: ex = exp(alpha - amax[dst]); den[dst] += ex (thread per (e,h))
// ---------------------------------------------------------------------------
__global__ void k_softmax_ex(const int* __restrict__ ei,
                             float* __restrict__ alpha,
                             const unsigned* __restrict__ amax,
                             float* __restrict__ den, int E, int H) {
  long i = (long)blockIdx.x * blockDim.x + threadIdx.x;
  if (i >= (long)E * H) return;
  int e = (int)(i / H);
  int h = (int)(i - (long)e * H);
  int dst = ei[E + e];
  float m  = dec_f32(amax[(size_t)dst * H + h]);
  float ex = __expf(alpha[i] - m);
  alpha[i] = ex;
  atomicAdd(&den[(size_t)dst * H + h], ex);
}

// ---------------------------------------------------------------------------
// edge pass 3: agg[dst] += (v[src] + emb) * ex/den[dst]  (one wave per edge)
// ---------------------------------------------------------------------------
template <int HW, int HEADS>
__global__ void k_edge_msg(const int* __restrict__ ei,
                           const float* __restrict__ eattr,
                           const float* __restrict__ We,
                           const float* __restrict__ be,
                           const float* __restrict__ v,
                           const float* __restrict__ alpha,
                           const float* __restrict__ den,
                           float* __restrict__ agg, int E) {
  const int CH = HW / 32;
  const int C  = HW / HEADS;
  int wave = blockIdx.x * (blockDim.x >> 5) + (threadIdx.x >> 5);
  int lane = threadIdx.x & 31;
  if (wave >= E) return;

  int src = ei[wave];
  int dst = ei[E + wave];
  int c0  = lane * CH;
  int h   = c0 / C;

  float ea = eattr[(size_t)wave * 32 + lane];
  float emb[CH];
#pragma unroll
  for (int i = 0; i < CH; ++i) emb[i] = be[c0 + i];
#pragma unroll
  for (int j = 0; j < 32; ++j) {
    float aj = __shfl(ea, j, 32);
#pragma unroll
    for (int i = 0; i < CH; ++i) emb[i] += aj * We[j * HW + c0 + i];
  }

  float a = alpha[(size_t)wave * HEADS + h] / den[(size_t)dst * HEADS + h];
#pragma unroll
  for (int i = 0; i < CH; ++i) {
    float m = (v[(size_t)src * HW + c0 + i] + emb[i]) * a;
    atomicAdd(&agg[(size_t)dst * HW + c0 + i], m);
  }
}

// ---------------------------------------------------------------------------
// node epilogue: h = LN( (relu?) (agg + skip) ) -> f32 + bf16 mirror
// one wave per node
// ---------------------------------------------------------------------------
template <int HW>
__global__ void k_node_ln(const float* __restrict__ agg,
                          const float* __restrict__ skip,
                          const float* __restrict__ gamma,
                          const float* __restrict__ beta,
                          float* __restrict__ outf,
                          unsigned short* __restrict__ outbf,
                          int Nn, int do_relu) {
  const int CH = HW / 32;
  int node = blockIdx.x * (blockDim.x >> 5) + (threadIdx.x >> 5);
  int lane = threadIdx.x & 31;
  if (node >= Nn) return;

  float vals[CH];
  float s = 0.0f;
#pragma unroll
  for (int i = 0; i < CH; ++i) {
    size_t idx = (size_t)node * HW + lane * CH + i;
    float t = agg[idx] + skip[idx];
    if (do_relu) t = fmaxf(t, 0.0f);
    vals[i] = t;
    s += t;
  }
#pragma unroll
  for (int off = 16; off > 0; off >>= 1) s += __shfl_xor(s, off, 32);
  float mean = s / (float)HW;

  float vs = 0.0f;
#pragma unroll
  for (int i = 0; i < CH; ++i) { float d = vals[i] - mean; vs += d * d; }
#pragma unroll
  for (int off = 16; off > 0; off >>= 1) vs += __shfl_xor(vs, off, 32);
  float inv = rsqrtf(vs / (float)HW + 1e-5f);

#pragma unroll
  for (int i = 0; i < CH; ++i) {
    int c = lane * CH + i;
    float o = (vals[i] - mean) * inv * gamma[c] + beta[c];
    size_t idx = (size_t)node * HW + c;
    outf[idx] = o;
    if (outbf) outbf[idx] = bf16_of(o);
  }
}

// ---------------------------------------------------------------------------
// host-side launcher
// ---------------------------------------------------------------------------
extern "C" void kernel_launch(void* const* d_in, const int* in_sizes, int n_in,
                              void* d_out, int out_size, void* d_ws,
                              size_t ws_size, hipStream_t stream) {
  const int IN = 512, PRE = 256, HID = 128, H = 2, OUT = 64, HW = H * HID; // 256
  const int Nn = in_sizes[0] / IN;     // 10000
  const int E  = in_sizes[1] / 2;      // 320000

  const float* x      = (const float*)d_in[0];
  const int*   ei     = (const int*)  d_in[1];
  const float* eattr  = (const float*)d_in[2];
  const float* pre_W  = (const float*)d_in[3];
  const float* pre_b  = (const float*)d_in[4];
  const float* c1_Wq  = (const float*)d_in[5];  const float* c1_bq = (const float*)d_in[6];
  const float* c1_Wk  = (const float*)d_in[7];  const float* c1_bk = (const float*)d_in[8];
  const float* c1_Wv  = (const float*)d_in[9];  const float* c1_bv = (const float*)d_in[10];
  const float* c1_We  = (const float*)d_in[11]; const float* c1_be = (const float*)d_in[12];
  const float* c1_Ws  = (const float*)d_in[13]; const float* c1_bs = (const float*)d_in[14];
  const float* n1_g   = (const float*)d_in[15]; const float* n1_b  = (const float*)d_in[16];
  const float* c2_Wq  = (const float*)d_in[17]; const float* c2_bq = (const float*)d_in[18];
  const float* c2_Wk  = (const float*)d_in[19]; const float* c2_bk = (const float*)d_in[20];
  const float* c2_Wv  = (const float*)d_in[21]; const float* c2_bv = (const float*)d_in[22];
  const float* c2_We  = (const float*)d_in[23]; const float* c2_be = (const float*)d_in[24];
  const float* c2_Ws  = (const float*)d_in[25]; const float* c2_bs = (const float*)d_in[26];
  const float* n2_g   = (const float*)d_in[27]; const float* n2_b  = (const float*)d_in[28];
  const float* mu_W   = (const float*)d_in[29]; const float* mu_b  = (const float*)d_in[30];
  const float* lv_W   = (const float*)d_in[31]; const float* lv_b  = (const float*)d_in[32];

  // ---- workspace carving (all offsets 256B aligned) ----
  char* ws = (char*)d_ws;
  size_t off = 0;
  auto alloc = [&](size_t bytes) {
    size_t o = off;
    off = (off + bytes + 255) & ~(size_t)255;
    return o;
  };
  unsigned short* xbf    = (unsigned short*)(ws + alloc((size_t)Nn * IN * 2));
  unsigned short* preWbf = (unsigned short*)(ws + alloc((size_t)IN * PRE * 2));
  unsigned short* W1bf[4];
  for (int i = 0; i < 4; ++i)
    W1bf[i] = (unsigned short*)(ws + alloc((size_t)PRE * HW * 2));
  unsigned short* W2bf[4];
  for (int i = 0; i < 4; ++i)
    W2bf[i] = (unsigned short*)(ws + alloc((size_t)HW * OUT * 2));
  unsigned short* muWbf = (unsigned short*)(ws + alloc((size_t)OUT * OUT * 2));
  unsigned short* lvWbf = (unsigned short*)(ws + alloc((size_t)OUT * OUT * 2));

  float*          h0f  = (float*)(ws + alloc((size_t)Nn * PRE * 4));
  unsigned short* h0bf = (unsigned short*)(ws + alloc((size_t)Nn * PRE * 2));
  float* q1 = (float*)(ws + alloc((size_t)Nn * HW * 4));
  float* k1 = (float*)(ws + alloc((size_t)Nn * HW * 4));
  float* v1 = (float*)(ws + alloc((size_t)Nn * HW * 4));
  float* s1 = (float*)(ws + alloc((size_t)Nn * HW * 4));
  float*    alpha1 = (float*)(ws + alloc((size_t)E * H * 4));
  unsigned* amax1  = (unsigned*)(ws + alloc((size_t)Nn * H * 4));
  float*    den1   = (float*)(ws + alloc((size_t)Nn * H * 4));
  float*    agg1   = (float*)(ws + alloc((size_t)Nn * HW * 4));
  float*          h1f  = (float*)(ws + alloc((size_t)Nn * HW * 4));
  unsigned short* h1bf = (unsigned short*)(ws + alloc((size_t)Nn * HW * 2));

  // conv2 buffers reuse conv1 scratch (all strictly smaller)
  float*    q2 = q1;   float* k2 = k1;   float* v2 = v1;   float* s2 = s1;
  float*    alpha2 = alpha1;
  unsigned* amax2  = amax1;
  float*    den2   = den1;
  float*    agg2   = agg1;
  float*          h2f  = h0f;
  unsigned short* h2bf = h0bf;

  float* out_mu = (float*)d_out;
  float* out_lv = out_mu + (size_t)Nn * OUT;

  // ---- helpers ----
  auto cvt = [&](const float* src, unsigned short* dst, long n) {
    int blk = 256;
    int grid = (int)((n + blk - 1) / blk);
    if (grid > 16384) grid = 16384;
    k_f32_to_bf16<<<grid, blk, 0, stream>>>(src, dst, n);
  };
  auto cvtT = [&](const float* W, unsigned short* Wt, int K, int Nc) {
    long n = (long)K * Nc;
    int blk = 256;
    int grid = (int)((n + blk - 1) / blk);
    if (grid > 16384) grid = 16384;
    k_transpose_bf16<<<grid, blk, 0, stream>>>(W, Wt, K, Nc);
  };
  auto fill = [&](void* p, unsigned val, long n) {
    int blk = 256;
    int grid = (int)((n + blk - 1) / blk);
    if (grid > 16384) grid = 16384;
    k_fill_u32<<<grid, blk, 0, stream>>>((unsigned*)p, val, n);
  };
  auto gemm = [&](const unsigned short* A, const unsigned short* Bt,
                  const float* bias, float* C, unsigned short* Cbf,
                  int M, int Nc, int K, int relu) {
    dim3 grid(Nc / 64, (M + 63) / 64);
    k_gemm_bf16<<<grid, 128, 0, stream>>>(A, Bt, bias, C, Cbf, M, Nc, K, relu);
  };
  const int EW_BLK = 256;                       // 8 waves/block
  const int ew_grid = (E + 7) / 8;

  // ---- 0) convert activations, transpose+convert weights to bf16 ----
  cvt(x, xbf, (long)Nn * IN);
  cvtT(pre_W, preWbf, IN, PRE);
  cvtT(c1_Wq, W1bf[0], PRE, HW);
  cvtT(c1_Wk, W1bf[1], PRE, HW);
  cvtT(c1_Wv, W1bf[2], PRE, HW);
  cvtT(c1_Ws, W1bf[3], PRE, HW);
  cvtT(c2_Wq, W2bf[0], HW, OUT);
  cvtT(c2_Wk, W2bf[1], HW, OUT);
  cvtT(c2_Wv, W2bf[2], HW, OUT);
  cvtT(c2_Ws, W2bf[3], HW, OUT);
  cvtT(mu_W, muWbf, OUT, OUT);
  cvtT(lv_W, lvWbf, OUT, OUT);

  // ---- 1) pre-MLP: h0 = relu(x @ pre_W + pre_b) ----
  gemm(xbf, preWbf, pre_b, h0f, h0bf, Nn, PRE, IN, 1);

  // ---- 2) conv1 projections ----
  gemm(h0bf, W1bf[0], c1_bq, q1, nullptr, Nn, HW, PRE, 0);
  gemm(h0bf, W1bf[1], c1_bk, k1, nullptr, Nn, HW, PRE, 0);
  gemm(h0bf, W1bf[2], c1_bv, v1, nullptr, Nn, HW, PRE, 0);
  gemm(h0bf, W1bf[3], c1_bs, s1, nullptr, Nn, HW, PRE, 0);

  // ---- 3) conv1 edge attention ----
  fill(amax1, ENC_NEG_INF, (long)Nn * H);
  fill(den1, 0u, (long)Nn * H);
  fill(agg1, 0u, (long)Nn * HW);
  k_edge_alpha<256, 2><<<ew_grid, EW_BLK, 0, stream>>>(
      ei, eattr, c1_We, c1_be, q1, k1, alpha1, amax1, E,
      0.08838834764831845f /* 1/sqrt(128) */);
  k_softmax_ex<<<((long)E * H + 255) / 256, 256, 0, stream>>>(
      ei, alpha1, amax1, den1, E, H);
  k_edge_msg<256, 2><<<ew_grid, EW_BLK, 0, stream>>>(
      ei, eattr, c1_We, c1_be, v1, alpha1, den1, agg1, E);

  // ---- 4) node epilogue 1: h1 = LN(relu(agg1 + skip1)) ----
  k_node_ln<256><<<(Nn + 7) / 8, 256, 0, stream>>>(
      agg1, s1, n1_g, n1_b, h1f, h1bf, Nn, 1);

  // ---- 5) conv2 projections ----
  gemm(h1bf, W2bf[0], c2_bq, q2, nullptr, Nn, OUT, HW, 0);
  gemm(h1bf, W2bf[1], c2_bk, k2, nullptr, Nn, OUT, HW, 0);
  gemm(h1bf, W2bf[2], c2_bv, v2, nullptr, Nn, OUT, HW, 0);
  gemm(h1bf, W2bf[3], c2_bs, s2, nullptr, Nn, OUT, HW, 0);

  // ---- 6) conv2 edge attention (heads=1, C=64) ----
  fill(amax2, ENC_NEG_INF, (long)Nn);
  fill(den2, 0u, (long)Nn);
  fill(agg2, 0u, (long)Nn * OUT);
  k_edge_alpha<64, 1><<<ew_grid, EW_BLK, 0, stream>>>(
      ei, eattr, c2_We, c2_be, q2, k2, alpha2, amax2, E,
      0.125f /* 1/sqrt(64) */);
  k_softmax_ex<<<((long)E + 255) / 256, 256, 0, stream>>>(
      ei, alpha2, amax2, den2, E, 1);
  k_edge_msg<64, 1><<<ew_grid, EW_BLK, 0, stream>>>(
      ei, eattr, c2_We, c2_be, v2, alpha2, den2, agg2, E);

  // ---- 7) node epilogue 2: h2 = LN(agg2 + skip2) ----
  k_node_ln<64><<<(Nn + 7) / 8, 256, 0, stream>>>(
      agg2, s2, n2_g, n2_b, h2f, h2bf, Nn, 0);

  // ---- 8) variational heads straight into d_out ----
  gemm(h2bf, muWbf, mu_b, out_mu, nullptr, Nn, OUT, OUT, 0);
  gemm(h2bf, lvWbf, lv_b, out_lv, nullptr, Nn, OUT, OUT, 0);
}
